// CustomGNN_16149077033388
// MI455X (gfx1250) — matrix operand order repaired and checked by reading, not compiled, not run on hardware
//
#include <hip/hip_runtime.h>

typedef _Float16 half_t;
typedef __attribute__((ext_vector_type(16))) _Float16 v16h;
typedef __attribute__((ext_vector_type(8)))  float    v8f;

// ---------------------------------------------------------------------------
// WMMA helpers (CDNA5 wave32, V_WMMA_F32_16X16X32_F16)
// A fragment: 16x32 f16, lane m=l&15, hi=l>>4: halves 0..7 -> K=kbase+hi*8+j,
//             halves 8..15 -> K=kbase+16+hi*8+j   (per ISA 16-bit A table)
// B fragment: 32x16 f16, lane n=l&15, hi=l>>4: halves 0..7 -> K=kbase+hi*8+j,
//             halves 8..15 -> K=kbase+16+hi*8+j   (row-major W[out][in])
// D fragment: 16x16 f32, lane n=l&15, vgpr r -> M = r + 8*(l>>4)
// ---------------------------------------------------------------------------
__device__ __forceinline__ v16h ldsA(const half_t* A, int stride, int mbase, int kbase) {
  int l = threadIdx.x & 31;
  int m = l & 15, hi = l >> 4;
  const half_t* p = A + (mbase + m) * stride + kbase + hi * 8;
  v16h a;
#pragma unroll
  for (int j = 0; j < 8; ++j) { a[j] = p[j]; a[j + 8] = p[j + 16]; }
  return a;
}

__device__ __forceinline__ v16h gldB(const half_t* W, int K, int nbase, int kbase) {
  int l = threadIdx.x & 31;
  int n = l & 15, hi = l >> 4;
  const half_t* p = W + (size_t)(nbase + n) * K + kbase + hi * 8;
  v16h b;
#pragma unroll
  for (int j = 0; j < 8; ++j) { b[j] = p[j]; b[j + 8] = p[j + 16]; }
  return b;
}

// out[m][n] = sum_k A[m][k] * W[n][k] + bias[n]   (optionally relu)
__device__ __forceinline__ void gemm_tiles(const half_t* A, int astride,
    const half_t* W, int K, const float* bias, float* out, int ostride,
    int Mt, int Nt, bool relu) {
  int lane = threadIdx.x & 31;
  int wave = threadIdx.x >> 5;
  int nwaves = blockDim.x >> 5;
  int ksteps = K >> 5;
  int n = lane & 15, hi = lane >> 4;
  for (int job = wave; job < Mt * Nt; job += nwaves) {
    int mt = job / Nt, nt = job - mt * Nt;
    v8f acc = {};
    for (int ks = 0; ks < ksteps; ++ks) {
      v16h a = ldsA(A, astride, mt * 16, ks * 32);
      v16h b = gldB(W, K, nt * 16, ks * 32);
      acc = __builtin_amdgcn_wmma_f32_16x16x32_f16(false, a, false, b,
                                                   (short)0, acc, false, false);
    }
    float bv = bias[nt * 16 + n];
#pragma unroll
    for (int r = 0; r < 8; ++r) {
      float v = acc[r] + bv;
      if (relu) v = fmaxf(v, 0.0f);
      out[(mt * 16 + hi * 8 + r) * ostride + nt * 16 + n] = v;
    }
  }
}

// LayerNorm over D=128: Xf[r][:] = LN(Xf[r][:] + add[r][:]) * g + b
__device__ __forceinline__ void layernorm_rows(float* Xf, const float* add, int addstride,
                                               const float* g, const float* b, int rows) {
  int tid = threadIdx.x;
  if (tid < rows) {
    float m = 0.f;
    for (int c = 0; c < 128; ++c) m += Xf[tid * 128 + c] + add[tid * addstride + c];
    m *= (1.0f / 128.0f);
    float v = 0.f;
    for (int c = 0; c < 128; ++c) {
      float d = Xf[tid * 128 + c] + add[tid * addstride + c] - m;
      v += d * d;
    }
    v *= (1.0f / 128.0f);
    float inv = rsqrtf(v + 1e-5f);
    for (int c = 0; c < 128; ++c) {
      float d = Xf[tid * 128 + c] + add[tid * addstride + c];
      Xf[tid * 128 + c] = (d - m) * inv * g[c] + b[c];
    }
  }
}

// ---------------------------------------------------------------------------
struct EncP {
  const half_t *in_w, *out_w, *lin1_w, *lin2_w;
  const float *in_b, *out_b, *ln1_g, *ln1_b, *lin1_b, *lin2_b, *ln2_g, *ln2_b;
};
struct FrontP {
  const float* w0; const float* b0;
  const half_t* w1; const float* b1;
  const half_t* w2; const float* b2;
};

// ---------------------------------------------------------------------------
// Front MLP: 10 -> 128 -> 128 -> 128 over 65536 nodes, level-0 override.
// blockIdx.y: 0 = hf -> hlogic, 1 = hs -> htrans. 64 rows / block.
// ---------------------------------------------------------------------------
__global__ void __launch_bounds__(256)
front_kernel(const float* __restrict__ x, const float* __restrict__ y_prob1,
             const float* __restrict__ y_trans_prob,
             FrontP Pf, FrontP Ps, float* hlogic, float* htrans) {
  extern __shared__ char smem[];
  float*  F0 = (float*)smem;                                 // 64*128 f32
  half_t* Hh = (half_t*)(smem + 64 * 128 * 4);               // 64*128 f16
  float*  F1 = (float*)(smem + 64 * 128 * 4 + 64 * 128 * 2); // 64*128 f32
  int tid = threadIdx.x;
  int type = blockIdx.y;
  FrontP Pp = type ? Ps : Pf;
  float* out = type ? htrans : hlogic;
  int rbase = blockIdx.x * 64;

  for (int o = tid; o < 64 * 128; o += 256) {
    int r = o >> 7, c = o & 127;
    float acc = Pp.b0[c];
    const float* xr = x + (size_t)(rbase + r) * 10;
    const float* wr = Pp.w0 + c * 10;
#pragma unroll
    for (int k = 0; k < 10; ++k) acc += xr[k] * wr[k];
    F0[o] = fmaxf(acc, 0.f);
  }
  __syncthreads();
  for (int o = tid; o < 64 * 128; o += 256) Hh[o] = (half_t)F0[o];
  __syncthreads();
  gemm_tiles(Hh, 128, Pp.w1, 128, Pp.b1, F1, 128, 4, 8, true);
  __syncthreads();
  for (int o = tid; o < 64 * 128; o += 256) Hh[o] = (half_t)F1[o];
  __syncthreads();
  gemm_tiles(Hh, 128, Pp.w2, 128, Pp.b2, F0, 128, 4, 8, false);
  __syncthreads();
  for (int o = tid; o < 64 * 128; o += 256) {
    int r = o >> 7, c = o & 127;
    int node = rbase + r;
    float v = F0[o];
    if (node < 4096) {                      // forward_level == 0
      if (type == 0) v = y_prob1[node];
      else {
        float s = 0.f;
#pragma unroll
        for (int j = 0; j < 8; ++j) s += y_trans_prob[node * 8 + j];
        v = s * 0.125f;
      }
    }
    out[(size_t)node * 128 + c] = v;
  }
}

// ---------------------------------------------------------------------------
// Propagate one level: 16 gates/block, 48 token rows, 2 encoder layers.
// blockIdx.y: 0 = AND params on hlogic, 1 = NOT params on htrans.
// ---------------------------------------------------------------------------
__global__ void __launch_bounds__(256)
propagate_kernel(float* hlogic, float* htrans, const int* __restrict__ idx_mat,
                 const int* __restrict__ tgt_idx, const float* __restrict__ pe,
                 EncP Pa, EncP Pn, int level) {
  extern __shared__ char smem[];
  float*  Xf   = (float*)smem;                                        // 48*128 f32
  half_t* Xh   = (half_t*)(smem + 48 * 128 * 4);                      // 48*384 f16
  float*  SC   = (float*)(smem + 48 * 128 * 4 + 48 * 384 * 2);        // 48*384 f32
  int*    meta = (int*)(smem + 48 * 128 * 4 + 48 * 384 * 2 + 48 * 384 * 4);

  int tid = threadIdx.x;
  int type = blockIdx.y;
  float* h = type ? htrans : hlogic;
  EncP Pp = type ? Pn : Pa;
  int gbase = blockIdx.x * 16;

  if (tid < 16) {
    int g = gbase + tid;
    int t  = tgt_idx[level * 4096 + g];
    int i0 = idx_mat[(size_t)(level * 4096 + g) * 2 + 0];
    int i1 = idx_mat[(size_t)(level * 4096 + g) * 2 + 1];
    meta[tid]      = t;
    meta[16 + tid] = (i0 < 0 ? 1 : 0) | (i1 < 0 ? 2 : 0);
    meta[32 + tid] = i0 < 0 ? 0 : i0;
    meta[48 + tid] = i1 < 0 ? 0 : i1;
  }
  __syncthreads();

  // gather sequences + positional encoding
  for (int e = tid; e < 48 * 128; e += 256) {
    int row = e >> 7, c = e & 127;
    int gi = row / 3, s = row - gi * 3;
    int node = (s == 0) ? meta[gi] : (s == 1 ? meta[32 + gi] : meta[48 + gi]);
    Xf[e] = h[(size_t)node * 128 + c] + pe[s * 128 + c];
  }
  __syncthreads();

  for (int li = 0; li < 2; ++li) {
    const half_t* in_w   = Pp.in_w   + li * 384 * 128;
    const half_t* out_w  = Pp.out_w  + li * 128 * 128;
    const half_t* lin1_w = Pp.lin1_w + li * 256 * 128;
    const half_t* lin2_w = Pp.lin2_w + li * 128 * 256;
    const float* in_b   = Pp.in_b   + li * 384;
    const float* out_b  = Pp.out_b  + li * 128;
    const float* ln1_g  = Pp.ln1_g  + li * 128;
    const float* ln1_b  = Pp.ln1_b  + li * 128;
    const float* lin1_b = Pp.lin1_b + li * 256;
    const float* lin2_b = Pp.lin2_b + li * 128;
    const float* ln2_g  = Pp.ln2_g  + li * 128;
    const float* ln2_b  = Pp.ln2_b  + li * 128;

    // x -> f16
    for (int e = tid; e < 48 * 128; e += 256)
      Xh[(e >> 7) * 384 + (e & 127)] = (half_t)Xf[e];
    __syncthreads();

    // QKV = x @ in_w^T + in_b   -> SC[48][384] (Q|K|V)
    gemm_tiles(Xh, 384, in_w, 128, in_b, SC, 384, 3, 24, false);
    __syncthreads();

    // attention per (gate, head); attn output -> Xh[:, 0:128] (f16)
    if (tid < 128) {
      int gi = tid >> 3, hh = tid & 7;
      int r0 = gi * 3;
      int padb = meta[16 + gi];
      for (int qs = 0; qs < 3; ++qs) {
        float sc[3];
#pragma unroll
        for (int ks = 0; ks < 3; ++ks) {
          float s = 0.f;
#pragma unroll
          for (int d = 0; d < 16; ++d)
            s += SC[(r0 + qs) * 384 + hh * 16 + d] *
                 SC[(r0 + ks) * 384 + 128 + hh * 16 + d];
          sc[ks] = s * 0.25f;
        }
        if (padb & 1) sc[1] = -1e9f;
        if (padb & 2) sc[2] = -1e9f;
        float mx = fmaxf(sc[0], fmaxf(sc[1], sc[2]));
        float e0 = expf(sc[0] - mx), e1 = expf(sc[1] - mx), e2 = expf(sc[2] - mx);
        float inv = 1.0f / (e0 + e1 + e2);
        e0 *= inv; e1 *= inv; e2 *= inv;
#pragma unroll
        for (int d = 0; d < 16; ++d) {
          float o = e0 * SC[(r0 + 0) * 384 + 256 + hh * 16 + d] +
                    e1 * SC[(r0 + 1) * 384 + 256 + hh * 16 + d] +
                    e2 * SC[(r0 + 2) * 384 + 256 + hh * 16 + d];
          Xh[(r0 + qs) * 384 + hh * 16 + d] = (half_t)o;
        }
      }
    }
    __syncthreads();

    // out-proj -> SC[:, 0:128]; then LN1(residual)
    gemm_tiles(Xh, 384, out_w, 128, out_b, SC, 384, 3, 8, false);
    __syncthreads();
    layernorm_rows(Xf, SC, 384, ln1_g, ln1_b, 48);
    __syncthreads();

    // FF: relu(x @ lin1^T + b) @ lin2^T + b ; then LN2(residual)
    for (int e = tid; e < 48 * 128; e += 256)
      Xh[(e >> 7) * 384 + (e & 127)] = (half_t)Xf[e];
    __syncthreads();
    gemm_tiles(Xh, 384, lin1_w, 128, lin1_b, SC, 384, 3, 16, true);
    __syncthreads();
    for (int e = tid; e < 48 * 256; e += 256)
      Xh[(e >> 8) * 384 + (e & 255)] = (half_t)SC[(e >> 8) * 384 + (e & 255)];
    __syncthreads();
    gemm_tiles(Xh, 384, lin2_w, 256, lin2_b, SC, 384, 3, 8, false);
    __syncthreads();
    layernorm_rows(Xf, SC, 384, ln2_g, ln2_b, 48);
    __syncthreads();
  }

  // scatter slot-0 rows back into h[tgt]
  for (int e = tid; e < 16 * 128; e += 256) {
    int gi = e >> 7, c = e & 127;
    h[(size_t)meta[gi] * 128 + c] = Xf[(gi * 3) * 128 + c];
  }
}

// ---------------------------------------------------------------------------
// Readout MLPs: rp(hlogic)->prob[1], rt(htrans)->trans[2]
// ---------------------------------------------------------------------------
__global__ void __launch_bounds__(256)
readout_kernel(const float* __restrict__ hlogic, const float* __restrict__ htrans,
               const float* rp_w0, const float* rp_b0, const float* rp_w1,
               const float* rp_b1, const float* rp_w2, const float* rp_b2,
               const float* rt_w0, const float* rt_b0, const float* rt_w1,
               const float* rt_b1, const float* rt_w2, const float* rt_b2,
               float* prob, float* trans) {
  __shared__ float W0f[2 * 32 * 128];
  int tid = threadIdx.x;
  for (int e = tid; e < 4096; e += 256) { W0f[e] = rp_w0[e]; W0f[4096 + e] = rt_w0[e]; }
  __syncthreads();
  int node = blockIdx.x * 256 + tid;

  {
    const float* hrow = hlogic + (size_t)node * 128;
    float h1[32], h2[32];
    for (int j = 0; j < 32; ++j) {
      float a = rp_b0[j];
      for (int k = 0; k < 128; ++k) a += hrow[k] * W0f[j * 128 + k];
      h1[j] = fmaxf(a, 0.f);
    }
    for (int j = 0; j < 32; ++j) {
      float a = rp_b1[j];
      for (int k = 0; k < 32; ++k) a += h1[k] * rp_w1[j * 32 + k];
      h2[j] = fmaxf(a, 0.f);
    }
    float a = rp_b2[0];
    for (int k = 0; k < 32; ++k) a += h2[k] * rp_w2[k];
    prob[node] = a;
  }
  {
    const float* hrow = htrans + (size_t)node * 128;
    float h1[32], h2[32];
    for (int j = 0; j < 32; ++j) {
      float a = rt_b0[j];
      for (int k = 0; k < 128; ++k) a += hrow[k] * W0f[4096 + j * 128 + k];
      h1[j] = fmaxf(a, 0.f);
    }
    for (int j = 0; j < 32; ++j) {
      float a = rt_b1[j];
      for (int k = 0; k < 32; ++k) a += h1[k] * rt_w1[j * 32 + k];
      h2[j] = fmaxf(a, 0.f);
    }
    for (int o = 0; o < 2; ++o) {
      float a = rt_b2[o];
      for (int k = 0; k < 32; ++k) a += h2[k] * rt_w2[o * 32 + k];
      trans[(size_t)node * 2 + o] = a;
    }
  }
}

__global__ void f2h_kernel(const float* __restrict__ src, half_t* __restrict__ dst, int n) {
  int i = blockIdx.x * blockDim.x + threadIdx.x;
  int stride = gridDim.x * blockDim.x;
  for (; i < n; i += stride) dst[i] = (half_t)src[i];
}

// ---------------------------------------------------------------------------
extern "C" void kernel_launch(void* const* d_in, const int* in_sizes, int n_in,
                              void* d_out, int out_size, void* d_ws, size_t ws_size,
                              hipStream_t stream) {
  (void)in_sizes; (void)n_in; (void)out_size; (void)ws_size;
  const int N = 65536;
  const size_t ND = (size_t)N * 128;

  const float* x            = (const float*)d_in[0];
  const float* y_prob1      = (const float*)d_in[2];
  const float* y_trans_prob = (const float*)d_in[3];
  const int*   idx_mat      = (const int*)d_in[4];
  const int*   tgt_idx      = (const int*)d_in[5];
  const float* pe           = (const float*)d_in[6];

  float* hlogic = (float*)d_out;
  float* htrans = hlogic + ND;
  float* zerosH = hlogic + 2 * ND;
  float* prob   = hlogic + 3 * ND;
  float* zerosP = prob + N;
  float* trans  = prob + 2 * N;

  // f16 weight arena in d_ws
  half_t* wsh = (half_t*)d_ws;
  half_t* whf1 = wsh + 0;        // 16384  (hf_w1)
  half_t* whf2 = wsh + 16384;    // 16384  (hf_w2)
  half_t* whs1 = wsh + 32768;    // 16384
  half_t* whs2 = wsh + 49152;    // 16384
  half_t* wa_in  = wsh + 65536;  // 98304  (and in_w, both layers)
  half_t* wa_out = wsh + 163840; // 32768
  half_t* wa_l1  = wsh + 196608; // 65536
  half_t* wa_l2  = wsh + 262144; // 65536
  half_t* wn_in  = wsh + 327680; // 98304
  half_t* wn_out = wsh + 425984; // 32768
  half_t* wn_l1  = wsh + 458752; // 65536
  half_t* wn_l2  = wsh + 524288; // 65536  -> end 589824 halves (~1.18 MB)

  auto conv = [&](const void* s, half_t* dptr, int n) {
    f2h_kernel<<<dim3((n + 1023) / 1024), dim3(256), 0, stream>>>((const float*)s, dptr, n);
  };
  conv(d_in[9],  whf1, 128 * 128);      // hf_w1
  conv(d_in[11], whf2, 128 * 128);      // hf_w2
  conv(d_in[15], whs1, 128 * 128);      // hs_w1
  conv(d_in[17], whs2, 128 * 128);      // hs_w2
  conv(d_in[31], wa_in,  2 * 384 * 128);
  conv(d_in[33], wa_out, 2 * 128 * 128);
  conv(d_in[37], wa_l1,  2 * 256 * 128);
  conv(d_in[39], wa_l2,  2 * 128 * 256);
  conv(d_in[43], wn_in,  2 * 384 * 128);
  conv(d_in[45], wn_out, 2 * 128 * 128);
  conv(d_in[49], wn_l1,  2 * 256 * 128);
  conv(d_in[51], wn_l2,  2 * 128 * 256);

  FrontP Pf { (const float*)d_in[7],  (const float*)d_in[8],  whf1,
              (const float*)d_in[10], whf2, (const float*)d_in[12] };
  FrontP Ps { (const float*)d_in[13], (const float*)d_in[14], whs1,
              (const float*)d_in[16], whs2, (const float*)d_in[18] };

  EncP Pa { wa_in, wa_out, wa_l1, wa_l2,
            (const float*)d_in[32], (const float*)d_in[34],
            (const float*)d_in[35], (const float*)d_in[36],
            (const float*)d_in[38], (const float*)d_in[40],
            (const float*)d_in[41], (const float*)d_in[42] };
  EncP Pn { wn_in, wn_out, wn_l1, wn_l2,
            (const float*)d_in[44], (const float*)d_in[46],
            (const float*)d_in[47], (const float*)d_in[48],
            (const float*)d_in[50], (const float*)d_in[52],
            (const float*)d_in[53], (const float*)d_in[54] };

  // Front MLPs (WMMA for the 128x128 layers)
  size_t flds = 64 * 128 * 4 + 64 * 128 * 2 + 64 * 128 * 4;  // 80 KB
  front_kernel<<<dim3(1024, 2), 256, flds, stream>>>(x, y_prob1, y_trans_prob,
                                                     Pf, Ps, hlogic, htrans);

  // 15 sequential propagation levels, AND & NOT encoders concurrently (grid.y)
  size_t plds = 48 * 128 * 4 + 48 * 384 * 2 + 48 * 384 * 4 + 64 * 4; // ~132 KB
  for (int l = 0; l < 15; ++l) {
    propagate_kernel<<<dim3(256, 2), 256, plds, stream>>>(hlogic, htrans, idx_mat,
                                                          tgt_idx, pe, Pa, Pn, l);
  }

  // Readouts
  readout_kernel<<<dim3(256), 256, 0, stream>>>(
      hlogic, htrans,
      (const float*)d_in[19], (const float*)d_in[20], (const float*)d_in[21],
      (const float*)d_in[22], (const float*)d_in[23], (const float*)d_in[24],
      (const float*)d_in[25], (const float*)d_in[26], (const float*)d_in[27],
      (const float*)d_in[28], (const float*)d_in[29], (const float*)d_in[30],
      prob, trans);

  // zeros_like outputs (graph-capturable memsets)
  hipMemsetAsync(zerosH, 0, ND * sizeof(float), stream);
  hipMemsetAsync(zerosP, 0, (size_t)N * sizeof(float), stream);
}
